// AttentionBlock_24429773979719
// MI455X (gfx1250) — matrix operand order repaired and checked by reading, not compiled, not run on hardware
//
#include <hip/hip_runtime.h>

typedef __attribute__((ext_vector_type(16))) _Float16 v16h;
typedef __attribute__((ext_vector_type(8)))  _Float16 v8h;
typedef __attribute__((ext_vector_type(4)))  _Float16 v4h;
typedef __attribute__((ext_vector_type(8)))  float    v8f;

#define B_   16
#define C_   512
#define N_   1024
#define G_   32
#define CG_  16
#define D_   512
#define EPS_ 1e-5f
#define SCALE_Q 0.044194173824159216f   // 512^-0.5

#define WMMA_F16(A, Bf, Cc) \
  __builtin_amdgcn_wmma_f32_16x16x32_f16(false, (A), false, (Bf), (short)0, (Cc), false, false)

// ---- WMMA fragment loaders (CDNA5 16x16x32 f16 layouts) ----
// A (16x32, MxK): lane l -> row (l&15); e -> K=kb+e, kb=(l>>4)*16
__device__ __forceinline__ v16h frag_a_f16(const _Float16* base, int stride, int lane){
  int row = lane & 15, kb = (lane >> 4) * 16;
  const _Float16* p = base + row * stride + kb;
  v8h lo = *(const v8h*)(p);
  v8h hi = *(const v8h*)(p + 8);
  v16h a;
#pragma unroll
  for (int i = 0; i < 8; ++i) { a[i] = lo[i]; a[i + 8] = hi[i]; }
  return a;
}
// B (32x16, KxN): lane l -> col (l&15); e -> K=(l>>4)*16 + e  => 16 contiguous f16 per lane
__device__ __forceinline__ v16h frag_b_f16(const _Float16* base, int stride, int lane){
  int col = lane & 15, kb = (lane >> 4) * 16;
  return *(const v16h*)(base + col * stride + kb);
}

// ---- Kernel 0: one-time f32 -> f16 weight conversion (L2-resident, removes cvt from hot loops)
__global__ void cvt_kernel(const float* __restrict__ src, _Float16* __restrict__ dst, int n4){
  int i = blockIdx.x * 256 + threadIdx.x;
  if (i < n4){
    float4 v = ((const float4*)src)[i];
    v4h o;
    o[0] = (_Float16)v.x; o[1] = (_Float16)v.y; o[2] = (_Float16)v.z; o[3] = (_Float16)v.w;
    ((v4h*)dst)[i] = o;
  }
}

// ---- Kernel 1: GroupNorm, write xnT[b][n][c] in f16 ----
__global__ void gn_kernel(const float* __restrict__ x, const float* __restrict__ gs,
                          const float* __restrict__ gb, _Float16* __restrict__ xnT){
  int bg = blockIdx.x;
  int b = bg / G_, g = bg % G_;
  const float* xp = x + ((size_t)b * C_ + (size_t)g * CG_) * N_;
  int t = threadIdx.x;
  const int TOT = CG_ * N_;            // 16384
  float s = 0.f, ss = 0.f;
  for (int i = t; i < TOT; i += 256){ float v = xp[i]; s += v; ss += v * v; }
  __shared__ float rs[256], rq[256];
  rs[t] = s; rq[t] = ss; __syncthreads();
  for (int o = 128; o > 0; o >>= 1){
    if (t < o){ rs[t] += rs[t + o]; rq[t] += rq[t + o]; }
    __syncthreads();
  }
  float mu  = rs[0] * (1.0f / TOT);
  float var = rq[0] * (1.0f / TOT) - mu * mu;
  float inv = rsqrtf(var + EPS_);
  for (int i = t; i < TOT; i += 256){
    int cl = i >> 10;          // i / N_
    int n  = i & (N_ - 1);
    int c  = g * CG_ + cl;
    float v = (xp[i] - mu) * inv * gs[c] + gb[c];
    xnT[((size_t)b * N_ + n) * C_ + c] = (_Float16)v;
  }
}

// ---- pipelined GEMM fragment set: 32 m-rows x 64 n-cols per wave, one 32-deep k-step ----
struct Frags { v16h a0, a1, b0, b1, b2, b3; };

__device__ __forceinline__ void load_frags(Frags& f, const _Float16* wb, const _Float16* xb,
                                           int k0, int lane){
  f.a0 = frag_a_f16(wb + k0,            C_, lane);
  f.a1 = frag_a_f16(wb + 16 * C_ + k0,  C_, lane);
  f.b0 = frag_b_f16(xb + 0 * 16 * C_ + k0, C_, lane);
  f.b1 = frag_b_f16(xb + 1 * 16 * C_ + k0, C_, lane);
  f.b2 = frag_b_f16(xb + 2 * 16 * C_ + k0, C_, lane);
  f.b3 = frag_b_f16(xb + 3 * 16 * C_ + k0, C_, lane);
}
__device__ __forceinline__ void mma_frags(const Frags& f, v8f (&acc)[2][4]){
  acc[0][0] = WMMA_F16(f.a0, f.b0, acc[0][0]);
  acc[0][1] = WMMA_F16(f.a0, f.b1, acc[0][1]);
  acc[0][2] = WMMA_F16(f.a0, f.b2, acc[0][2]);
  acc[0][3] = WMMA_F16(f.a0, f.b3, acc[0][3]);
  acc[1][0] = WMMA_F16(f.a1, f.b0, acc[1][0]);
  acc[1][1] = WMMA_F16(f.a1, f.b1, acc[1][1]);
  acc[1][2] = WMMA_F16(f.a1, f.b2, acc[1][2]);
  acc[1][3] = WMMA_F16(f.a1, f.b3, acc[1][3]);
}

// ---- Kernel 2: QKV GEMM (3c x c) @ (c x n); outputs qT/kT [b][n][d], v [b][d][n] ----
__global__ void qkv_kernel(const _Float16* __restrict__ wh, const float* __restrict__ bias,
                           const _Float16* __restrict__ xnT,
                           _Float16* __restrict__ qT, _Float16* __restrict__ kT,
                           _Float16* __restrict__ vh){
  int lane = threadIdx.x & 31, wv = threadIdx.x >> 5;
  int n0 = blockIdx.x * 64, m0 = blockIdx.y * 256 + wv * 32, b = blockIdx.z;
  const _Float16* wb = wh  + (size_t)m0 * C_;
  const _Float16* xb = xnT + ((size_t)b * N_ + n0) * C_;
  v8f acc[2][4] = {};
  Frags f0, f1;
  load_frags(f0, wb, xb, 0, lane);
  for (int k0 = 0; k0 < C_; k0 += 64){
    load_frags(f1, wb, xb, k0 + 32, lane);
    mma_frags(f0, acc);
    load_frags(f0, wb, xb, (k0 + 64) & (C_ - 1), lane);   // wraps harmlessly on last step
    mma_frags(f1, acc);
  }
  int rbase = (lane >> 4) * 8, col = lane & 15;
#pragma unroll
  for (int mt = 0; mt < 2; ++mt){
#pragma unroll
    for (int nt = 0; nt < 4; ++nt){
      int n = n0 + nt * 16 + col;
#pragma unroll
      for (int r = 0; r < 8; ++r){
        int m = m0 + mt * 16 + rbase + r;
        float val = acc[mt][nt][r] + bias[m];
        if (m < C_)            qT[((size_t)b * N_ + n) * D_ + m]            = (_Float16)(val * SCALE_Q);
        else if (m < 2 * C_)   kT[((size_t)b * N_ + n) * D_ + (m - C_)]     = (_Float16)val;
        else                   vh[((size_t)b * D_ + (m - 2 * C_)) * N_ + n] = (_Float16)val;
      }
    }
  }
}

// ---- S = Q(16x512) . K^T tile (16 j-cols), pipelined B-frag stream from L2 ----
__device__ __forceinline__ v8f qk_tile(const _Float16* Qs, const _Float16* krow, int lane){
  v8f acc = {};
  v16h bcur = frag_b_f16(krow, D_, lane);
#pragma unroll
  for (int d0 = 0; d0 < D_; d0 += 32){
    v16h a = frag_a_f16(Qs + d0, D_, lane);
    v16h bn;
    if (d0 + 32 < D_) bn = frag_b_f16(krow + d0 + 32, D_, lane);
    acc = WMMA_F16(a, bcur, acc);
    if (d0 + 32 < D_) bcur = bn;
  }
  return acc;
}

// ---- Kernel 3: two-pass flash attention, one (batch, 16-query tile) per workgroup ----
__global__ void attn_kernel(const _Float16* __restrict__ qT, const _Float16* __restrict__ kT,
                            const _Float16* __restrict__ vv, _Float16* __restrict__ oT){
  int lane = threadIdx.x & 31, wv = threadIdx.x >> 5;
  int i0 = blockIdx.x * 16, b = blockIdx.y;
  __shared__ _Float16 Qs[16 * 512];     // 16 KB, [m][d]
  __shared__ _Float16 Ps[16 * 128];     // 4 KB,  [m][j]
  __shared__ float smax[8][16], ssum[8][16], Mrow[16], Lrec[16];

  { // stage Q tile (contiguous 16 KB)
    const v8h* src = (const v8h*)(qT + ((size_t)b * N_ + i0) * D_);
    v8h* dst = (v8h*)Qs;
    for (int i = threadIdx.x; i < (16 * 512) / 8; i += 256) dst[i] = src[i];
  }
  __syncthreads();

  int rbase = (lane >> 4) * 8, col = lane & 15;
  const _Float16* kb = kT + (size_t)b * N_ * D_;
  const _Float16* vb = vv + (size_t)b * D_ * N_;

  // ---- Pass 1: online softmax statistics; wave wv owns j-columns j0 + wv*16 .. +15
  float mrun[8], lrun[8];
#pragma unroll
  for (int r = 0; r < 8; ++r){ mrun[r] = -3.0e38f; lrun[r] = 0.f; }

  for (int j0 = 0; j0 < N_; j0 += 128){
    if (j0 + 128 < N_)
      __builtin_prefetch(kb + (size_t)(j0 + 128 + wv * 16 + col) * D_, 0, 0);
    v8f acc = qk_tile(Qs, kb + (size_t)(j0 + wv * 16) * D_, lane);
#pragma unroll
    for (int r = 0; r < 8; ++r){
      float s = acc[r];
      float rm = s;
#pragma unroll
      for (int msk = 1; msk < 16; msk <<= 1) rm = fmaxf(rm, __shfl_xor(rm, msk, 32));
      float mn = fmaxf(mrun[r], rm);
      float p = __expf(s - mn);
      float rsum = p;
#pragma unroll
      for (int msk = 1; msk < 16; msk <<= 1) rsum += __shfl_xor(rsum, msk, 32);
      lrun[r] = lrun[r] * __expf(mrun[r] - mn) + rsum;
      mrun[r] = mn;
    }
  }
  if (col == 0){
#pragma unroll
    for (int r = 0; r < 8; ++r){ smax[wv][rbase + r] = mrun[r]; ssum[wv][rbase + r] = lrun[r]; }
  }
  __syncthreads();
  if (threadIdx.x < 16){
    int row = threadIdx.x;
    float M = -3.0e38f;
    for (int w2 = 0; w2 < 8; ++w2) M = fmaxf(M, smax[w2][row]);
    float L = 0.f;
    for (int w2 = 0; w2 < 8; ++w2) L += ssum[w2][row] * __expf(smax[w2][row] - M);
    Mrow[row] = M;
    Lrec[row] = 1.0f / L;
  }
  __syncthreads();

  float Mr[8], Lr[8];
#pragma unroll
  for (int r = 0; r < 8; ++r){ Mr[r] = Mrow[rbase + r]; Lr[r] = Lrec[rbase + r]; }

  // ---- Pass 2: recompute S -> P into LDS, then P @ V^T; wave wv owns d-slice wv*64..+63
  v8f accO[4] = {};
  for (int j0 = 0; j0 < N_; j0 += 128){
    v8f acc = qk_tile(Qs, kb + (size_t)(j0 + wv * 16) * D_, lane);
#pragma unroll
    for (int r = 0; r < 8; ++r){
      float p = __expf(acc[r] - Mr[r]);
      Ps[(rbase + r) * 128 + wv * 16 + col] = (_Float16)p;
    }
    __syncthreads();
#pragma unroll
    for (int kt = 0; kt < 4; ++kt){
      v16h a = frag_a_f16(Ps + kt * 32, 128, lane);
#pragma unroll
      for (int dt = 0; dt < 4; ++dt){
        v16h bf = frag_b_f16(vb + (size_t)(wv * 64 + dt * 16) * N_ + j0 + kt * 32, N_, lane);
        accO[dt] = WMMA_F16(a, bf, accO[dt]);
      }
    }
    __syncthreads();
  }

#pragma unroll
  for (int dt = 0; dt < 4; ++dt){
    int d = wv * 64 + dt * 16 + col;
#pragma unroll
    for (int r = 0; r < 8; ++r){
      int q = i0 + rbase + r;
      oT[((size_t)b * N_ + q) * D_ + d] = (_Float16)(accO[dt][r] * Lr[r]);
    }
  }
}

// ---- Kernel 4: proj GEMM + bias + residual (f32 out) ----
__global__ void proj_kernel(const _Float16* __restrict__ wh, const float* __restrict__ bias,
                            const _Float16* __restrict__ oT, const float* __restrict__ x,
                            float* __restrict__ out){
  int lane = threadIdx.x & 31, wv = threadIdx.x >> 5;
  int n0 = blockIdx.x * 64, m0 = blockIdx.y * 256 + wv * 32, b = blockIdx.z;
  const _Float16* wb = wh + (size_t)m0 * C_;
  const _Float16* ob = oT + ((size_t)b * N_ + n0) * D_;   // stride D_ == C_ == 512
  v8f acc[2][4] = {};
  Frags f0, f1;
  load_frags(f0, wb, ob, 0, lane);
  for (int k0 = 0; k0 < C_; k0 += 64){
    load_frags(f1, wb, ob, k0 + 32, lane);
    mma_frags(f0, acc);
    load_frags(f0, wb, ob, (k0 + 64) & (C_ - 1), lane);
    mma_frags(f1, acc);
  }
  int rbase = (lane >> 4) * 8, col = lane & 15;
#pragma unroll
  for (int mt = 0; mt < 2; ++mt){
#pragma unroll
    for (int nt = 0; nt < 4; ++nt){
      int n = n0 + nt * 16 + col;
#pragma unroll
      for (int r = 0; r < 8; ++r){
        int m = m0 + mt * 16 + rbase + r;
        size_t idx = ((size_t)b * C_ + m) * N_ + n;
        out[idx] = x[idx] + acc[mt][nt][r] + bias[m];
      }
    }
  }
}

extern "C" void kernel_launch(void* const* d_in, const int* in_sizes, int n_in,
                              void* d_out, int out_size, void* d_ws, size_t ws_size,
                              hipStream_t stream) {
  (void)in_sizes; (void)n_in; (void)out_size; (void)ws_size;
  const float* x      = (const float*)d_in[0];
  const float* gn_s   = (const float*)d_in[1];
  const float* gn_b   = (const float*)d_in[2];
  const float* qkv_w  = (const float*)d_in[3];
  const float* qkv_b  = (const float*)d_in[4];
  const float* proj_w = (const float*)d_in[5];
  const float* proj_b = (const float*)d_in[6];
  float* out = (float*)d_out;

  char* ws = (char*)d_ws;
  const size_t SZ = (size_t)B_ * N_ * C_ * sizeof(_Float16);   // 16 MB per buffer
  _Float16* xnT = (_Float16*)(ws);
  _Float16* qT  = (_Float16*)(ws + 1 * SZ);
  _Float16* kT  = (_Float16*)(ws + 2 * SZ);
  _Float16* vh  = (_Float16*)(ws + 3 * SZ);
  _Float16* oT  = (_Float16*)(ws + 4 * SZ);
  _Float16* wqh = (_Float16*)(ws + 5 * SZ);                              // 1.5 MB
  _Float16* wph = (_Float16*)(ws + 5 * SZ + (size_t)3 * C_ * C_ * 2);    // 0.5 MB

  const int nq4 = (3 * C_ * C_) / 4, np4 = (C_ * C_) / 4;
  cvt_kernel <<<dim3((nq4 + 255) / 256), dim3(256), 0, stream>>>(qkv_w,  wqh, nq4);
  cvt_kernel <<<dim3((np4 + 255) / 256), dim3(256), 0, stream>>>(proj_w, wph, np4);

  gn_kernel  <<<dim3(B_ * G_),                 dim3(256), 0, stream>>>(x, gn_s, gn_b, xnT);
  qkv_kernel <<<dim3(N_/64, (3*C_)/256, B_),   dim3(256), 0, stream>>>(wqh, qkv_b, xnT, qT, kT, vh);
  attn_kernel<<<dim3(N_/16, B_),               dim3(256), 0, stream>>>(qT, kT, vh, oT);
  proj_kernel<<<dim3(N_/64, C_/256, B_),       dim3(256), 0, stream>>>(wph, proj_b, oT, x, out);
}
